// Block_39195871544051
// MI455X (gfx1250) — compile-verified, matrix operand order checked by reading, not run on hardware
//
#include <hip/hip_runtime.h>
#include <hip/hip_bf16.h>
#include <math.h>

// ---------------- types ----------------
typedef __attribute__((ext_vector_type(16))) __bf16 v16bf;
typedef __attribute__((ext_vector_type(8)))  float  v8f;

__device__ __forceinline__ v8f wmma_bf16(v16bf a, v16bf b, v8f c) {
  return __builtin_amdgcn_wmma_f32_16x16x32_bf16(false, a, false, b, (short)0, c,
                                                 false, false);
}

// async copy 16B/lane from global (saddr + voffset) into LDS at per-lane addr
#define ASYNC_B128(ldsaddr, voff, sbase)                                        \
  asm volatile("global_load_async_to_lds_b128 %0, %1, %2"                        \
               :: "v"(ldsaddr), "v"(voff),                                       \
                  "s"((unsigned long long)(uintptr_t)(sbase)) : "memory")
#define WAIT_ASYNC0() asm volatile("s_wait_asynccnt 0" ::: "memory")
#define WAIT_ASYNC4() asm volatile("s_wait_asynccnt 4" ::: "memory")
#define WAIT_DS()     asm volatile("s_wait_dscnt 0" ::: "memory")

#define BT   2
#define TSEQ 2048
#define CDIM 1024
#define NH   16
#define HSZ  64
#define MROWS (BT * TSEQ)   // 4096

// ---------------- weight convert + transpose: W[K][N] f32 -> Wt[N][K] bf16 ---
__global__ __launch_bounds__(256) void wconv_kernel(const float* __restrict__ W,
                                                    __bf16* __restrict__ Wt,
                                                    int K, int N) {
  __shared__ float tile[64][65];
  const int k0 = blockIdx.x * 64, n0 = blockIdx.y * 64;
  const int c = threadIdx.x & 63, r4 = threadIdx.x >> 6;
#pragma unroll
  for (int i = 0; i < 16; ++i) {
    int r = r4 + i * 4;
    tile[r][c] = W[(size_t)(k0 + r) * N + n0 + c];
  }
  __syncthreads();
#pragma unroll
  for (int i = 0; i < 16; ++i) {
    int r = r4 + i * 4;  // n index
    Wt[(size_t)(n0 + r) * K + k0 + c] = (__bf16)tile[c][r];
  }
}

// ---------------- LayerNorm (one row per block), bf16 output ----------------
__global__ __launch_bounds__(256) void ln_kernel(const float* __restrict__ x,
                                                 const float* __restrict__ g,
                                                 const float* __restrict__ bta,
                                                 __bf16* __restrict__ out) {
  const int row = blockIdx.x;
  const int tid = threadIdx.x;
  const float* xr = x + (size_t)row * CDIM;
  float v[4];
  float s = 0.f;
#pragma unroll
  for (int i = 0; i < 4; ++i) { v[i] = xr[tid + i * 256]; s += v[i]; }
  __shared__ float red[256];
  red[tid] = s;
  __syncthreads();
#pragma unroll
  for (int off = 128; off > 0; off >>= 1) {
    if (tid < off) red[tid] += red[tid + off];
    __syncthreads();
  }
  const float mu = red[0] * (1.0f / CDIM);
  __syncthreads();
  float s2 = 0.f;
#pragma unroll
  for (int i = 0; i < 4; ++i) { float d = v[i] - mu; s2 += d * d; }
  red[tid] = s2;
  __syncthreads();
#pragma unroll
  for (int off = 128; off > 0; off >>= 1) {
    if (tid < off) red[tid] += red[tid + off];
    __syncthreads();
  }
  const float rstd = rsqrtf(red[0] * (1.0f / CDIM) + 1e-5f);
#pragma unroll
  for (int i = 0; i < 4; ++i) {
    int c = tid + i * 256;
    out[(size_t)row * CDIM + c] = (__bf16)((v[i] - mu) * rstd * g[c] + bta[c]);
  }
}

// ------- bf16 WMMA GEMM: out[M,N] = A[M,K] @ Bt[N,K]^T (+epilogue) ----------
// A, Bt are bf16; Bt is pre-transposed [N][K].  Double-buffered async staging.
// EPI: 0 = +bias, 1 = +bias then exact GELU, 2 = +bias +residual(f32)
// OBF: true -> bf16 output, false -> f32 output
template <int EPI, bool OBF>
__global__ __launch_bounds__(256) void gemm_bf16(const __bf16* __restrict__ A,
                                                 const __bf16* __restrict__ Bt_g,
                                                 const float* __restrict__ bias,
                                                 const float* __restrict__ res,
                                                 void* __restrict__ outv,
                                                 int K, int N) {
  __shared__ __align__(16) __bf16 As[2][128][40];  // [m][k], pad 8 bf16
  __shared__ __align__(16) __bf16 Bs[2][128][40];  // [n][k], pad 8 bf16
  const int tid = threadIdx.x;
  const int lane = tid & 31;
  const int wave = tid >> 5;
  const int laneN = lane & 15;
  const int half = lane >> 4;
  const int wr = wave >> 2;            // 0..1 -> 64-row band
  const int wc = wave & 3;             // 0..3 -> 32-col band
  const int m0 = blockIdx.y * 128;
  const int n0 = blockIdx.x * 128;

  // 512 chunks of 16B per tile; this thread owns chunks tid and tid+256
  const int r0 = tid >> 2, c8 = tid & 3;  // row, 16B-chunk within 32-elem row
  const int r1 = r0 + 64;
  const unsigned voff0 = (unsigned)(r0 * K + c8 * 8) * 2u;
  const unsigned voff1 = (unsigned)(r1 * K + c8 * 8) * 2u;
  unsigned lA0[2], lA1[2], lB0[2], lB1[2];
#pragma unroll
  for (int p = 0; p < 2; ++p) {
    lA0[p] = (unsigned)(uintptr_t)&As[p][r0][c8 * 8];
    lA1[p] = (unsigned)(uintptr_t)&As[p][r1][c8 * 8];
    lB0[p] = (unsigned)(uintptr_t)&Bs[p][r0][c8 * 8];
    lB1[p] = (unsigned)(uintptr_t)&Bs[p][r1][c8 * 8];
  }
  const __bf16* aBase = A + (size_t)m0 * K;
  const __bf16* bBase = Bt_g + (size_t)n0 * K;

  v8f acc[4][2];
#pragma unroll
  for (int mi = 0; mi < 4; ++mi)
#pragma unroll
    for (int ni = 0; ni < 2; ++ni) acc[mi][ni] = (v8f)0.0f;

  // prologue: stage tile 0 into buffer 0
  ASYNC_B128(lA0[0], voff0, aBase);
  ASYNC_B128(lA1[0], voff1, aBase);
  ASYNC_B128(lB0[0], voff0, bBase);
  ASYNC_B128(lB1[0], voff1, bBase);

  const int KT = K >> 5;
  for (int kt = 0; kt < KT; ++kt) {
    const int p = kt & 1, q = p ^ 1;
    if (kt + 1 < KT) {
      // prefetch next tile into the other buffer; it stays in flight
      // through this iteration's WMMAs (async loads complete in order).
      const __bf16* sa = aBase + (kt + 1) * 32;
      const __bf16* sb = bBase + (kt + 1) * 32;
      ASYNC_B128(lA0[q], voff0, sa);
      ASYNC_B128(lA1[q], voff1, sa);
      ASYNC_B128(lB0[q], voff0, sb);
      ASYNC_B128(lB1[q], voff1, sb);
      WAIT_ASYNC4();   // oldest 4 (current tile) done; newest 4 outstanding
    } else {
      WAIT_ASYNC0();
    }
    __syncthreads();

    v16bf af[4], bfr[2];
#pragma unroll
    for (int mi = 0; mi < 4; ++mi)
#pragma unroll
      for (int j = 0; j < 16; ++j)
        af[mi][j] = As[p][wr * 64 + mi * 16 + laneN][16 * (j >> 3) + 8 * half + (j & 7)];
#pragma unroll
    for (int ni = 0; ni < 2; ++ni)
#pragma unroll
      for (int j = 0; j < 16; ++j)
        bfr[ni][j] = Bs[p][wc * 32 + ni * 16 + laneN][16 * (j >> 3) + 8 * half + (j & 7)];
#pragma unroll
    for (int mi = 0; mi < 4; ++mi)
#pragma unroll
      for (int ni = 0; ni < 2; ++ni)
        acc[mi][ni] = wmma_bf16(af[mi], bfr[ni], acc[mi][ni]);
    __syncthreads();
  }

#pragma unroll
  for (int mi = 0; mi < 4; ++mi)
#pragma unroll
    for (int ni = 0; ni < 2; ++ni)
#pragma unroll
      for (int v = 0; v < 8; ++v) {
        int row = m0 + wr * 64 + mi * 16 + v + 8 * half;
        int col = n0 + wc * 32 + ni * 16 + laneN;
        float val = acc[mi][ni][v] + bias[col];
        if (EPI == 1) val = 0.5f * val * (1.0f + erff(val * 0.70710678118654752f));
        if (EPI == 2) val += res[(size_t)row * N + col];
        if (OBF) ((__bf16*)outv)[(size_t)row * N + col] = (__bf16)val;
        else     ((float*)outv)[(size_t)row * N + col] = val;
      }
}

// ---------------- causal flash attention, one wave per 32-query tile --------
__global__ __launch_bounds__(32) void attn_kernel(const __bf16* __restrict__ qkv,
                                                  __bf16* __restrict__ out) {
  const int QT = TSEQ / 32;
  const int qt = blockIdx.x % QT;
  const int bh = blockIdx.x / QT;
  const int h = bh % NH;
  const int b = bh / NH;
  const int lane = threadIdx.x;
  const int laneN = lane & 15;
  const int half = lane >> 4;
  const int q0 = qt * 32;
  const int RS = 3 * CDIM;

  __shared__ __align__(16) __bf16 Qs[32][HSZ + 8];
  __shared__ __align__(16) __bf16 Ks[32][HSZ + 8];
  __shared__ __align__(16) __bf16 Vt[HSZ][32 + 8];
  __shared__ __align__(16) __bf16 Ps[32][32 + 8];

  // async-load Q tile (32 x 64 bf16, 256 chunks of 16B)
  const __bf16* qbase = qkv + (size_t)(b * TSEQ + q0) * RS + (CDIM + h * HSZ);
#pragma unroll
  for (int it = 0; it < 8; ++it) {
    int c = lane + it * 32;
    int r = c >> 3, c8 = c & 7;
    unsigned voff = (unsigned)(r * RS + c8 * 8) * 2u;
    unsigned lds = (unsigned)(uintptr_t)&Qs[r][c8 * 8];
    ASYNC_B128(lds, voff, qbase);
  }

  v8f o[2][4];
  float mrow[2][8], lrow[2][8];
#pragma unroll
  for (int mi = 0; mi < 2; ++mi) {
#pragma unroll
    for (int di = 0; di < 4; ++di) o[mi][di] = (v8f)0.0f;
#pragma unroll
    for (int v = 0; v < 8; ++v) { mrow[mi][v] = -__builtin_inff(); lrow[mi][v] = 0.f; }
  }

  const int nkt = qt + 1;
  for (int kt = 0; kt < nkt; ++kt) {
    const int k0 = kt * 32;
    const __bf16* kbase = qkv + (size_t)(b * TSEQ + k0) * RS + (h * HSZ);
    const __bf16* vbase = qkv + (size_t)(b * TSEQ + k0) * RS + (2 * CDIM + h * HSZ);
    WAIT_DS();   // prior-iter frag ds_loads done before async overwrites Ks
#pragma unroll
    for (int it = 0; it < 8; ++it) {
      int c = lane + it * 32;
      int r = c >> 3, c8 = c & 7;
      unsigned voff = (unsigned)(r * RS + c8 * 8) * 2u;
      unsigned lds = (unsigned)(uintptr_t)&Ks[r][c8 * 8];
      ASYNC_B128(lds, voff, kbase);
    }
    // V transposed into LDS (manual; async cannot transpose)
    for (int i = lane; i < 32 * HSZ; i += 32) {
      int r = i >> 6, c = i & 63;
      Vt[c][r] = vbase[(size_t)r * RS + c];
    }
    WAIT_ASYNC0();
    WAIT_DS();

    // S = Q @ K^T  (32q x 32k), K-reduction over HS=64 in two steps of 32
    v8f s[2][2];
#pragma unroll
    for (int mi = 0; mi < 2; ++mi)
#pragma unroll
      for (int ni = 0; ni < 2; ++ni) s[mi][ni] = (v8f)0.0f;
#pragma unroll
    for (int ks = 0; ks < 2; ++ks) {
      v16bf aq[2], bk[2];
#pragma unroll
      for (int mi = 0; mi < 2; ++mi)
#pragma unroll
        for (int j = 0; j < 16; ++j)
          aq[mi][j] = Qs[mi * 16 + laneN][32 * ks + 16 * (j >> 3) + 8 * half + (j & 7)];
#pragma unroll
      for (int ni = 0; ni < 2; ++ni)
#pragma unroll
        for (int j = 0; j < 16; ++j)
          bk[ni][j] = Ks[ni * 16 + laneN][32 * ks + 16 * (j >> 3) + 8 * half + (j & 7)];
#pragma unroll
      for (int mi = 0; mi < 2; ++mi)
#pragma unroll
        for (int ni = 0; ni < 2; ++ni)
          s[mi][ni] = wmma_bf16(aq[mi], bk[ni], s[mi][ni]);
    }

    // mask + online softmax (row = v + 8*half in C layout; 16 lanes = 16 cols)
#pragma unroll
    for (int mi = 0; mi < 2; ++mi) {
#pragma unroll
      for (int v = 0; v < 8; ++v) {
        const int q = q0 + mi * 16 + v + 8 * half;
        float s0 = (k0 + laneN <= q)      ? s[mi][0][v] * 0.125f : -__builtin_inff();
        float s1 = (k0 + 16 + laneN <= q) ? s[mi][1][v] * 0.125f : -__builtin_inff();
        float mx = fmaxf(s0, s1);
#pragma unroll
        for (int off = 1; off < 16; off <<= 1) mx = fmaxf(mx, __shfl_xor(mx, off, 16));
        const float mnew = fmaxf(mrow[mi][v], mx);
        const float fac = __expf(mrow[mi][v] - mnew);
        const float p0 = __expf(s0 - mnew);
        const float p1 = __expf(s1 - mnew);
        float rs = p0 + p1;
#pragma unroll
        for (int off = 1; off < 16; off <<= 1) rs += __shfl_xor(rs, off, 16);
        lrow[mi][v] = lrow[mi][v] * fac + rs;
        mrow[mi][v] = mnew;
#pragma unroll
        for (int di = 0; di < 4; ++di) o[mi][di][v] *= fac;
        Ps[mi * 16 + v + 8 * half][laneN] = (__bf16)p0;
        Ps[mi * 16 + v + 8 * half][16 + laneN] = (__bf16)p1;
      }
    }
    WAIT_DS();

    // O += P @ V
    v16bf ap[2], bv[4];
#pragma unroll
    for (int mi = 0; mi < 2; ++mi)
#pragma unroll
      for (int j = 0; j < 16; ++j)
        ap[mi][j] = Ps[mi * 16 + laneN][16 * (j >> 3) + 8 * half + (j & 7)];
#pragma unroll
    for (int di = 0; di < 4; ++di)
#pragma unroll
      for (int j = 0; j < 16; ++j)
        bv[di][j] = Vt[di * 16 + laneN][16 * (j >> 3) + 8 * half + (j & 7)];
#pragma unroll
    for (int mi = 0; mi < 2; ++mi)
#pragma unroll
      for (int di = 0; di < 4; ++di)
        o[mi][di] = wmma_bf16(ap[mi], bv[di], o[mi][di]);
  }

  // write [B,T,C] with head concat (bf16)
#pragma unroll
  for (int mi = 0; mi < 2; ++mi)
#pragma unroll
    for (int di = 0; di < 4; ++di)
#pragma unroll
      for (int v = 0; v < 8; ++v) {
        const int q = q0 + mi * 16 + v + 8 * half;
        const int c = h * HSZ + di * 16 + laneN;
        out[(size_t)(b * TSEQ + q) * CDIM + c] = (__bf16)(o[mi][di][v] / lrow[mi][v]);
      }
}

// ---------------- host-side orchestration ----------------
extern "C" void kernel_launch(void* const* d_in, const int* in_sizes, int n_in,
                              void* d_out, int out_size, void* d_ws, size_t ws_size,
                              hipStream_t stream) {
  const float* x     = (const float*)d_in[0];
  const float* W_qkv = (const float*)d_in[1];
  const float* b_qkv = (const float*)d_in[2];
  const float* W_o   = (const float*)d_in[3];
  const float* b_o   = (const float*)d_in[4];
  const float* ln1_g = (const float*)d_in[5];
  const float* ln1_b = (const float*)d_in[6];
  const float* ln2_g = (const float*)d_in[7];
  const float* ln2_b = (const float*)d_in[8];
  const float* W1    = (const float*)d_in[9];
  const float* b1    = (const float*)d_in[10];
  const float* W2    = (const float*)d_in[11];
  const float* b2    = (const float*)d_in[12];
  float* out = (float*)d_out;

  float* x1 = (float*)d_ws;                               // 4096*1024 f32
  __bf16* bfp = (__bf16*)(x1 + (size_t)MROWS * CDIM);
  __bf16* wqkv_t = bfp;  bfp += (size_t)(3 * CDIM) * CDIM;     // [3C][C]
  __bf16* wo_t   = bfp;  bfp += (size_t)CDIM * CDIM;           // [C][C]
  __bf16* w1_t   = bfp;  bfp += (size_t)(4 * CDIM) * CDIM;     // [4C][C]
  __bf16* w2_t   = bfp;  bfp += (size_t)CDIM * (4 * CDIM);     // [C][4C]
  __bf16* hbuf   = bfp;  bfp += (size_t)MROWS * CDIM;
  __bf16* qkvb   = bfp;  bfp += (size_t)MROWS * 3 * CDIM;
  __bf16* attb   = bfp;  bfp += (size_t)MROWS * CDIM;
  __bf16* ff1    = bfp;  bfp += (size_t)MROWS * 4 * CDIM;

  // 0) weights -> bf16, transposed to [N][K]
  wconv_kernel<<<dim3(CDIM / 64, 3 * CDIM / 64), 256, 0, stream>>>(W_qkv, wqkv_t, CDIM, 3 * CDIM);
  wconv_kernel<<<dim3(CDIM / 64, CDIM / 64),     256, 0, stream>>>(W_o,   wo_t,   CDIM, CDIM);
  wconv_kernel<<<dim3(CDIM / 64, 4 * CDIM / 64), 256, 0, stream>>>(W1,    w1_t,   CDIM, 4 * CDIM);
  wconv_kernel<<<dim3(4 * CDIM / 64, CDIM / 64), 256, 0, stream>>>(W2,    w2_t,   4 * CDIM, CDIM);
  // 1) h = LN1(x)  (bf16)
  ln_kernel<<<MROWS, 256, 0, stream>>>(x, ln1_g, ln1_b, hbuf);
  // 2) qkv = h @ W_qkv + b_qkv  (bf16)
  gemm_bf16<0, true><<<dim3(3 * CDIM / 128, MROWS / 128), 256, 0, stream>>>(
      hbuf, wqkv_t, b_qkv, nullptr, qkvb, CDIM, 3 * CDIM);
  // 3) attn = causal_softmax(QK^T/sqrt(d)) @ V  (bf16)
  attn_kernel<<<BT * NH * (TSEQ / 32), 32, 0, stream>>>(qkvb, attb);
  // 4) x1 = x + attn @ W_o + b_o  (f32)
  gemm_bf16<2, false><<<dim3(CDIM / 128, MROWS / 128), 256, 0, stream>>>(
      attb, wo_t, b_o, x, x1, CDIM, CDIM);
  // 5) h = LN2(x1)  (bf16)
  ln_kernel<<<MROWS, 256, 0, stream>>>(x1, ln2_g, ln2_b, hbuf);
  // 6) ff1 = gelu(h @ W1 + b1)  (bf16)
  gemm_bf16<1, true><<<dim3(4 * CDIM / 128, MROWS / 128), 256, 0, stream>>>(
      hbuf, w1_t, b1, nullptr, ff1, CDIM, 4 * CDIM);
  // 7) out = x1 + ff1 @ W2 + b2  (f32)
  gemm_bf16<2, false><<<dim3(CDIM / 128, MROWS / 128), 256, 0, stream>>>(
      ff1, w2_t, b2, x1, out, 4 * CDIM, CDIM);
}
